// MultiHeadAttention_52278341927254
// MI455X (gfx1250) — matrix operand order repaired and checked by reading, not compiled, not run on hardware
//
#include <hip/hip_runtime.h>
#include <hip/hip_bf16.h>

#define D_MODEL 1024
#define N_HEAD  16
#define D_HEAD  64
#define SEQLEN  2048
#define NBATCH  4
#define NROWS   (NBATCH * SEQLEN)   // 8192

typedef __attribute__((ext_vector_type(16))) __bf16 v16bf;
typedef __attribute__((ext_vector_type(8)))  float  v8f;

union AFrag {
    v16bf v;
    uint4 q[2];
    unsigned short u[16];
};

__device__ __forceinline__ unsigned short f2bf(float f) {
    unsigned int u = __float_as_uint(f);
    u += 0x7FFFu + ((u >> 16) & 1u);   // round-to-nearest-even
    return (unsigned short)(u >> 16);
}

__device__ __forceinline__ v8f wmma_bf16(const AFrag& a, const AFrag& b, v8f c) {
    return __builtin_amdgcn_wmma_f32_16x16x32_bf16(false, a.v, false, b.v,
                                                   (short)0, c, false, false);
}

// ---------------- elementwise fp32 -> bf16 ----------------
__global__ __launch_bounds__(256) void conv_bf16_kernel(const float* __restrict__ src,
                                                        unsigned short* __restrict__ dst, int n) {
    int i = blockIdx.x * blockDim.x + threadIdx.x;
    int stride = gridDim.x * blockDim.x;
    for (; i < n; i += stride) dst[i] = f2bf(src[i]);
}

// W [K=1024][N=1024] fp32 -> Wt [N][K] bf16
__global__ __launch_bounds__(256) void transpose_bf16_kernel(const float* __restrict__ src,
                                                             unsigned short* __restrict__ dst) {
    int i = blockIdx.x * blockDim.x + threadIdx.x;   // over N*K, exact
    int n = i >> 10;
    int k = i & 1023;
    dst[i] = f2bf(src[k * 1024 + n]);
}

// ---------------- fused QKV projection GEMM ----------------
// Block tile 128x64, 4 waves, each wave 64x32 (8 WMMA per 12 b128 loads per K-step).
__global__ __launch_bounds__(128) void qkv_gemm_kernel(
    const unsigned short* __restrict__ xb,
    const unsigned short* __restrict__ wqt,
    const unsigned short* __restrict__ wkt,
    const unsigned short* __restrict__ wvt,
    const float* __restrict__ bq, const float* __restrict__ bk, const float* __restrict__ bv,
    unsigned short* __restrict__ qb, unsigned short* __restrict__ kb, unsigned short* __restrict__ vt)
{
    const int z = blockIdx.z;
    const unsigned short* wt = (z == 0) ? wqt : (z == 1) ? wkt : wvt;
    const float* bias = (z == 0) ? bq : (z == 1) ? bk : bv;

    const int m0 = blockIdx.x * 128;
    const int n0 = blockIdx.y * 64;
    const int lane = threadIdx.x & 31;
    const int wid = threadIdx.x >> 5;
    const int wm = wid & 1;       // 64-row half
    const int wn = wid >> 1;      // 32-col half
    const int col = lane & 15;
    const int hi = lane >> 4;

    v8f acc[4][2] = {};
    const int arow0 = m0 + wm * 64 + col;
    const int brow0 = n0 + wn * 32 + col;

#pragma unroll 2
    for (int k0 = 0; k0 < D_MODEL; k0 += 32) {
        AFrag a[4], bm[2];
#pragma unroll
        for (int am = 0; am < 4; ++am) {
            const unsigned short* p = xb + (size_t)(arow0 + am * 16) * D_MODEL + k0 + 8 * hi;
            a[am].q[0] = *(const uint4*)p;
            a[am].q[1] = *(const uint4*)(p + 16);
        }
#pragma unroll
        for (int bn = 0; bn < 2; ++bn) {
            const unsigned short* p = wt + (size_t)(brow0 + bn * 16) * D_MODEL + k0 + 8 * hi;
            bm[bn].q[0] = *(const uint4*)p;
            bm[bn].q[1] = *(const uint4*)(p + 16);
        }
#pragma unroll
        for (int am = 0; am < 4; ++am)
#pragma unroll
            for (int bn = 0; bn < 2; ++bn)
                acc[am][bn] = wmma_bf16(a[am], bm[bn], acc[am][bn]);
    }

    float bb[2];
    bb[0] = bias[n0 + wn * 32 + col];
    bb[1] = bias[n0 + wn * 32 + 16 + col];

    if (z < 2) {
        unsigned short* out = (z == 0) ? qb : kb;
#pragma unroll
        for (int am = 0; am < 4; ++am)
#pragma unroll
            for (int bn = 0; bn < 2; ++bn) {
                const int n = n0 + wn * 32 + bn * 16 + col;
                const int h = n >> 6, dh = n & 63;
#pragma unroll
                for (int j = 0; j < 8; ++j) {
                    const int m = m0 + wm * 64 + am * 16 + 8 * hi + j;
                    const int bi = m >> 11, s = m & (SEQLEN - 1);
                    out[((size_t)(bi * N_HEAD + h) * SEQLEN + s) * D_HEAD + dh] =
                        f2bf(acc[am][bn][j] + bb[bn]);
                }
            }
    } else {  // V stored transposed per head: vt[(bh*D_HEAD + dh)*SEQLEN + s]
#pragma unroll
        for (int am = 0; am < 4; ++am)
#pragma unroll
            for (int bn = 0; bn < 2; ++bn) {
                const int n = n0 + wn * 32 + bn * 16 + col;
                const int h = n >> 6, dh = n & 63;
                const int mbase = m0 + wm * 64 + am * 16 + 8 * hi;
                const int bi = mbase >> 11, sbase = mbase & (SEQLEN - 1);
                unsigned short* dst =
                    vt + ((size_t)(bi * N_HEAD + h) * D_HEAD + dh) * SEQLEN + sbase;
#pragma unroll
                for (int jj = 0; jj < 4; ++jj) {
                    unsigned int lo = f2bf(acc[am][bn][2 * jj] + bb[bn]);
                    unsigned int hh = f2bf(acc[am][bn][2 * jj + 1] + bb[bn]);
                    *(unsigned int*)(dst + 2 * jj) = lo | (hh << 16);
                }
            }
    }
}

// ---------------- flash attention per (b,h), 64 query rows / block ----------------
__global__ __launch_bounds__(128) void attn_kernel(
    const unsigned short* __restrict__ qb,
    const unsigned short* __restrict__ kb,
    const unsigned short* __restrict__ vt,
    unsigned short* __restrict__ attb)
{
    const int bh = blockIdx.y;
    const int b = bh >> 4, h = bh & 15;
    const int q0 = blockIdx.x * 64;
    const int lane = threadIdx.x & 31;
    const int wid = threadIdx.x >> 5;   // 0..3, each wave owns 16 query rows
    const int col = lane & 15;
    const int hi = lane >> 4;

    const unsigned short* Q = qb + (size_t)bh * SEQLEN * D_HEAD;
    const unsigned short* K = kb + (size_t)bh * SEQLEN * D_HEAD;
    const unsigned short* V = vt + (size_t)bh * D_HEAD * SEQLEN;

    __shared__ __align__(16) unsigned short plds[4][16][72];  // 16B-aligned rows, conflict-free

    AFrag qa[2];
#pragma unroll
    for (int ks = 0; ks < 2; ++ks) {
        const unsigned short* p = Q + (size_t)(q0 + wid * 16 + col) * D_HEAD + ks * 32 + 8 * hi;
        qa[ks].q[0] = *(const uint4*)p;
        qa[ks].q[1] = *(const uint4*)(p + 16);
    }

    float mrow[8], lrow[8];
#pragma unroll
    for (int j = 0; j < 8; ++j) { mrow[j] = -3.0e38f; lrow[j] = 0.0f; }
    v8f o[4] = {};

    for (int kt = 0; kt < SEQLEN; kt += 64) {
        // prefetch next K/V tiles (global_prefetch_b8) to overlap with softmax VALU work
        if (kt + 64 < SEQLEN) {
            __builtin_prefetch(K + (size_t)(kt + 64 + lane) * D_HEAD, 0, 0);
            __builtin_prefetch(K + (size_t)(kt + 96 + lane) * D_HEAD, 0, 0);
            __builtin_prefetch(V + (size_t)lane * SEQLEN + kt + 64, 0, 0);
            __builtin_prefetch(V + (size_t)(lane + 32) * SEQLEN + kt + 64, 0, 0);
        }

        // scores = Q Kt  (16 x 64 per wave)
        v8f sf[4];
#pragma unroll
        for (int g = 0; g < 4; ++g) {
            v8f s = {};
#pragma unroll
            for (int ks = 0; ks < 2; ++ks) {
                AFrag kf;
                const unsigned short* p =
                    K + (size_t)(kt + g * 16 + col) * D_HEAD + ks * 32 + 8 * hi;
                kf.q[0] = *(const uint4*)p;
                kf.q[1] = *(const uint4*)(p + 16);
                s = wmma_bf16(qa[ks], kf, s);
            }
            sf[g] = s * 0.125f;   // 1/sqrt(64)
        }

        // online softmax, per row j (16-lane reductions stay inside half-wave)
#pragma unroll
        for (int j = 0; j < 8; ++j) {
            float mx = fmaxf(fmaxf(sf[0][j], sf[1][j]), fmaxf(sf[2][j], sf[3][j]));
#pragma unroll
            for (int off = 8; off > 0; off >>= 1)
                mx = fmaxf(mx, __shfl_xor(mx, off, 32));
            const float mnew = fmaxf(mrow[j], mx);
            const float corr = __expf(mrow[j] - mnew);
            float sum = 0.0f;
#pragma unroll
            for (int g = 0; g < 4; ++g) {
                const float pv = __expf(sf[g][j] - mnew);
                sf[g][j] = pv;
                sum += pv;
            }
#pragma unroll
            for (int off = 8; off > 0; off >>= 1)
                sum += __shfl_xor(sum, off, 32);
            lrow[j] = lrow[j] * corr + sum;
            mrow[j] = mnew;
#pragma unroll
            for (int g = 0; g < 4; ++g) o[g][j] *= corr;
        }

        // relayout P (C-layout -> A-layout) through per-wave LDS region
#pragma unroll
        for (int g = 0; g < 4; ++g)
#pragma unroll
            for (int j = 0; j < 8; ++j)
                plds[wid][j + 8 * hi][g * 16 + col] = f2bf(sf[g][j]);

        asm volatile("s_wait_dscnt 0" ::: "memory");   // wave-private region; DS in-order per wave

        AFrag pa[2];
#pragma unroll
        for (int ks = 0; ks < 2; ++ks) {
            const unsigned short* p = &plds[wid][col][ks * 32 + 8 * hi];
            pa[ks].q[0] = *(const uint4*)p;
            pa[ks].q[1] = *(const uint4*)(p + 16);
        }

        // O += P V   (V transposed layout -> contiguous B fragments)
#pragma unroll
        for (int g = 0; g < 4; ++g) {
#pragma unroll
            for (int ks = 0; ks < 2; ++ks) {
                AFrag vf;
                const unsigned short* p =
                    V + (size_t)(g * 16 + col) * SEQLEN + kt + ks * 32 + 8 * hi;
                vf.q[0] = *(const uint4*)p;
                vf.q[1] = *(const uint4*)(p + 16);
                o[g] = wmma_bf16(pa[ks], vf, o[g]);
            }
        }
    }

    // normalize and store merged-head activations [B,S,D] bf16
#pragma unroll
    for (int j = 0; j < 8; ++j) {
        const float inv = 1.0f / lrow[j];
        const int s = q0 + wid * 16 + 8 * hi + j;
#pragma unroll
        for (int g = 0; g < 4; ++g) {
            const int n = h * D_HEAD + g * 16 + col;
            attb[(size_t)(b * SEQLEN + s) * D_MODEL + n] = f2bf(o[g][j] * inv);
        }
    }
}

// ---------------- output projection GEMM (fp32 out) ----------------
__global__ __launch_bounds__(128) void out_gemm_kernel(
    const unsigned short* __restrict__ ab,
    const unsigned short* __restrict__ wot,
    const float* __restrict__ bo,
    float* __restrict__ out)
{
    const int m0 = blockIdx.x * 128;
    const int n0 = blockIdx.y * 64;
    const int lane = threadIdx.x & 31;
    const int wid = threadIdx.x >> 5;
    const int wm = wid & 1, wn = wid >> 1;
    const int col = lane & 15;
    const int hi = lane >> 4;

    v8f acc[4][2] = {};
    const int arow0 = m0 + wm * 64 + col;
    const int brow0 = n0 + wn * 32 + col;

#pragma unroll 2
    for (int k0 = 0; k0 < D_MODEL; k0 += 32) {
        AFrag a[4], bm[2];
#pragma unroll
        for (int am = 0; am < 4; ++am) {
            const unsigned short* p = ab + (size_t)(arow0 + am * 16) * D_MODEL + k0 + 8 * hi;
            a[am].q[0] = *(const uint4*)p;
            a[am].q[1] = *(const uint4*)(p + 16);
        }
#pragma unroll
        for (int bn = 0; bn < 2; ++bn) {
            const unsigned short* p = wot + (size_t)(brow0 + bn * 16) * D_MODEL + k0 + 8 * hi;
            bm[bn].q[0] = *(const uint4*)p;
            bm[bn].q[1] = *(const uint4*)(p + 16);
        }
#pragma unroll
        for (int am = 0; am < 4; ++am)
#pragma unroll
            for (int bn = 0; bn < 2; ++bn)
                acc[am][bn] = wmma_bf16(a[am], bm[bn], acc[am][bn]);
    }

    float bb[2];
    bb[0] = bo[n0 + wn * 32 + col];
    bb[1] = bo[n0 + wn * 32 + 16 + col];

#pragma unroll
    for (int am = 0; am < 4; ++am)
#pragma unroll
        for (int bn = 0; bn < 2; ++bn) {
            const int n = n0 + wn * 32 + bn * 16 + col;
#pragma unroll
            for (int j = 0; j < 8; ++j) {
                const int m = m0 + wm * 64 + am * 16 + 8 * hi + j;
                out[(size_t)m * D_MODEL + n] = acc[am][bn][j] + bb[bn];
            }
        }
}

extern "C" void kernel_launch(void* const* d_in, const int* in_sizes, int n_in,
                              void* d_out, int out_size, void* d_ws, size_t ws_size,
                              hipStream_t stream) {
    (void)in_sizes; (void)n_in; (void)out_size; (void)ws_size;
    const float* x  = (const float*)d_in[0];
    const float* wq = (const float*)d_in[1];
    const float* bq = (const float*)d_in[2];
    const float* wk = (const float*)d_in[3];
    const float* bk = (const float*)d_in[4];
    const float* wv = (const float*)d_in[5];
    const float* bv = (const float*)d_in[6];
    const float* wo = (const float*)d_in[7];
    const float* bo = (const float*)d_in[8];
    float* out = (float*)d_out;

    char* ws = (char*)d_ws;
    const size_t MB = 1024 * 1024;
    unsigned short* xb   = (unsigned short*)(ws);              // 16 MB, reused as attb later
    unsigned short* wqt  = (unsigned short*)(ws + 16 * MB);    // 2 MB each
    unsigned short* wkt  = (unsigned short*)(ws + 18 * MB);
    unsigned short* wvt  = (unsigned short*)(ws + 20 * MB);
    unsigned short* wot  = (unsigned short*)(ws + 22 * MB);
    unsigned short* qb   = (unsigned short*)(ws + 24 * MB);    // 16 MB
    unsigned short* kb   = (unsigned short*)(ws + 40 * MB);    // 16 MB
    unsigned short* vt   = (unsigned short*)(ws + 56 * MB);    // 16 MB
    unsigned short* attb = xb;                                 // alias: xb dead after projections

    conv_bf16_kernel<<<4096, 256, 0, stream>>>(x, xb, NROWS * D_MODEL);
    transpose_bf16_kernel<<<4096, 256, 0, stream>>>(wq, wqt);
    transpose_bf16_kernel<<<4096, 256, 0, stream>>>(wk, wkt);
    transpose_bf16_kernel<<<4096, 256, 0, stream>>>(wv, wvt);
    transpose_bf16_kernel<<<4096, 256, 0, stream>>>(wo, wot);

    qkv_gemm_kernel<<<dim3(NROWS / 128, D_MODEL / 64, 3), 128, 0, stream>>>(
        xb, wqt, wkt, wvt, bq, bk, bv, qb, kb, vt);

    attn_kernel<<<dim3(SEQLEN / 64, NBATCH * N_HEAD), 128, 0, stream>>>(qb, kb, vt, attb);

    out_gemm_kernel<<<dim3(NROWS / 128, D_MODEL / 64), 128, 0, stream>>>(attb, wot, bo, out);
}
